// MAB_37890201485827
// MI455X (gfx1250) — matrix operand order repaired and checked by reading, not compiled, not run on hardware
//
#include <hip/hip_runtime.h>
#include <hip/hip_bf16.h>

typedef __bf16 v16bf __attribute__((ext_vector_type(16)));
typedef float  v8f   __attribute__((ext_vector_type(8)));
typedef unsigned int v8u __attribute__((ext_vector_type(8)));
typedef float  f32x4 __attribute__((ext_vector_type(4)));
typedef unsigned int u32x4 __attribute__((ext_vector_type(4)));

#define DV 512
#define SEQ 1024
#define NHEAD 8
#define DH 64

__device__ __forceinline__ unsigned short f2bf(float f) {
  unsigned int u = __builtin_bit_cast(unsigned int, f);
  unsigned int r = (u + 0x7FFFu + ((u >> 16) & 1u)) >> 16;
  return (unsigned short)r;
}
__device__ __forceinline__ float bf2f(unsigned short h) {
  unsigned int u = ((unsigned int)h) << 16;
  return __builtin_bit_cast(float, u);
}
__device__ __forceinline__ unsigned int pack2(float lo, float hi) {
  return (unsigned int)f2bf(lo) | ((unsigned int)f2bf(hi) << 16);
}

// ---- operand loaders (pure bf16) -----------------------------------------
// A operand (16x32 bf16): lane = row (m = lane&15); K per lane:
//   lanes 0-15: K {0..7, 16..23}; lanes 16-31: K {8..15, 24..31}
__device__ __forceinline__ v16bf loadA_bf16(const unsigned short* rowp, int lane) {
  const unsigned short* p = rowp + ((lane >> 4) << 3);
  u32x4 lo = *(const u32x4*)(p);
  u32x4 hi = *(const u32x4*)(p + 16);
  v8u u;
  u[0] = lo[0]; u[1] = lo[1]; u[2] = lo[2]; u[3] = lo[3];
  u[4] = hi[0]; u[5] = hi[1]; u[6] = hi[2]; u[7] = hi[3];
  return __builtin_bit_cast(v16bf, u);
}
// B operand (32x16 bf16): lane = column (n = lane&15); K per lane:
//   lanes 0-15: K {0..7, 16..23}; lanes 16-31: K {8..15, 24..31}
__device__ __forceinline__ v16bf loadB_bf16(const unsigned short* rowp, int lane) {
  const unsigned short* p = rowp + ((lane >> 4) << 3);
  u32x4 lo = *(const u32x4*)(p);
  u32x4 hi = *(const u32x4*)(p + 16);
  v8u u;
  u[0] = lo[0]; u[1] = lo[1]; u[2] = lo[2]; u[3] = lo[3];
  u[4] = hi[0]; u[5] = hi[1]; u[6] = hi[2]; u[7] = hi[3];
  return __builtin_bit_cast(v16bf, u);
}
__device__ __forceinline__ v8f wmma_bf16(v16bf a, v16bf b, v8f c) {
  return __builtin_amdgcn_wmma_f32_16x16x32_bf16(false, a, false, b, (short)0, c,
                                                 false, false);
}

// ---- async global->LDS staging (gfx1250, ASYNCcnt-tracked) ----------------
__device__ __forceinline__ void async_g2l_b128(unsigned lds_off,
                                               const unsigned short* gaddr) {
  asm volatile("global_load_async_to_lds_b128 %0, %1, off"
               :
               : "v"(lds_off), "v"(gaddr)
               : "memory");
}
__device__ __forceinline__ void wait_async0() {
  asm volatile("s_wait_asynccnt 0x0" ::: "memory");
}

// ---- elementwise fp32 -> bf16 conversion (vectorized, grid-stride) --------
__global__ void __launch_bounds__(256)
mab_cvt_bf16(const float* __restrict__ X, unsigned short* __restrict__ Y, int n8) {
  int stride = gridDim.x * blockDim.x;
  for (int i = blockIdx.x * blockDim.x + threadIdx.x; i < n8; i += stride) {
    const float* p = X + (size_t)i * 8;
    f32x4 a = *(const f32x4*)(p);
    f32x4 b = *(const f32x4*)(p + 4);
    u32x4 o;
    o[0] = pack2(a[0], a[1]); o[1] = pack2(a[2], a[3]);
    o[2] = pack2(b[0], b[1]); o[3] = pack2(b[2], b[3]);
    *(u32x4*)(Y + (size_t)i * 8) = o;
  }
}

// ---- projection GEMM: out = A(bf16) @ W(bf16)^T + bias, stored bf16 -------
// 32x64 tile per wave. trans==0: out row-major [M,DV]; trans==1: out [B,DV,SEQ]
__global__ void __launch_bounds__(256)
mab_gemm_proj(const unsigned short* __restrict__ A, const unsigned short* __restrict__ W,
              const float* __restrict__ bias, unsigned short* __restrict__ out,
              int M, int trans) {
  int lane = threadIdx.x & 31, wave = threadIdx.x >> 5;
  int wid = blockIdx.x * 8 + wave;
  const int ntiles = DV / 64;  // 8
  int m0 = (wid / ntiles) * 32, n0 = (wid % ntiles) * 64;
  if (m0 >= M) return;
  int ln = lane & 15, lh = lane >> 4;
  v8f c[2][4];
#pragma unroll
  for (int i = 0; i < 2; ++i)
#pragma unroll
    for (int t = 0; t < 4; ++t) c[i][t] = {};
  for (int k0 = 0; k0 < DV; k0 += 32) {
    v16bf a0 = loadA_bf16(A + (size_t)(m0 + ln) * DV + k0, lane);
    v16bf a1 = loadA_bf16(A + (size_t)(m0 + 16 + ln) * DV + k0, lane);
    v16bf b0 = loadB_bf16(W + (size_t)(n0 + ln) * DV + k0, lane);
    v16bf b1 = loadB_bf16(W + (size_t)(n0 + 16 + ln) * DV + k0, lane);
    v16bf b2 = loadB_bf16(W + (size_t)(n0 + 32 + ln) * DV + k0, lane);
    v16bf b3 = loadB_bf16(W + (size_t)(n0 + 48 + ln) * DV + k0, lane);
    c[0][0] = wmma_bf16(a0, b0, c[0][0]);
    c[0][1] = wmma_bf16(a0, b1, c[0][1]);
    c[0][2] = wmma_bf16(a0, b2, c[0][2]);
    c[0][3] = wmma_bf16(a0, b3, c[0][3]);
    c[1][0] = wmma_bf16(a1, b0, c[1][0]);
    c[1][1] = wmma_bf16(a1, b1, c[1][1]);
    c[1][2] = wmma_bf16(a1, b2, c[1][2]);
    c[1][3] = wmma_bf16(a1, b3, c[1][3]);
  }
#pragma unroll
  for (int i = 0; i < 2; ++i) {
#pragma unroll
    for (int t = 0; t < 4; ++t) {
      int mb = m0 + i * 16, nb = n0 + t * 16 + ln;
      float bi = bias[nb];
#pragma unroll
      for (int j = 0; j < 8; ++j) {
        int m = mb + j + lh * 8;
        float v = c[i][t][j] + bi;
        size_t idx;
        if (!trans) idx = (size_t)m * DV + nb;
        else {
          int bb = m >> 10, s = m & (SEQ - 1);
          idx = ((size_t)bb * DV + nb) * SEQ + s;
        }
        out[idx] = f2bf(v);
      }
    }
  }
}

// ---- flash attention + residual ------------------------------------------
// Qp,Kp bf16 [B*SEQ, DV]; VpT bf16 [B, DV, SEQ]; Opre f32 [B*SEQ, DV]
// 8 waves share (b,h): K/V chunks staged once per block into LDS via async
// loads (double buffered), WMMA operands fed from LDS.
__global__ void __launch_bounds__(256)
mab_attn(const unsigned short* __restrict__ Qp, const unsigned short* __restrict__ Kp,
         const unsigned short* __restrict__ VpT, float* __restrict__ Opre) {
  __shared__ unsigned short Kbuf[2][32 * 64];   // [buf][key][feat]  4KB each
  __shared__ unsigned short Vbuf[2][64 * 32];   // [buf][feat][key]  4KB each
  __shared__ unsigned short plds[8 * 16 * 32];  // per-wave 16x32 repack tile
  int tid = threadIdx.x;
  int lane = tid & 31, wave = tid >> 5;
  int qb = blockIdx.x & 7;
  int bh = blockIdx.x >> 3;
  int b = bh >> 3, h = bh & 7;
  int q0 = qb * 128 + wave * 16;
  int ln = lane & 15, lh = lane >> 4;
  const unsigned short* Qb = Qp + (size_t)b * SEQ * DV + h * DH;
  const unsigned short* Kb = Kp + (size_t)b * SEQ * DV + h * DH;
  const unsigned short* Vb = VpT + ((size_t)b * DV + h * DH) * SEQ;
  unsigned short* myl = plds + wave * 512;

  // per-thread staging slots: one b128 (8 bf16) each for K and V
  int krow = tid >> 3, kseg = tid & 7;          // K: 32 rows x 8 segs
  int vrow = tid >> 2, vseg = tid & 3;          // V: 64 rows x 4 segs
  const unsigned short* kg = Kb + (size_t)krow * DV + kseg * 8;
  const unsigned short* vg = Vb + (size_t)vrow * SEQ + vseg * 8;
  unsigned klo[2], vlo[2];
#pragma unroll
  for (int u = 0; u < 2; ++u) {
    klo[u] = (unsigned)(uintptr_t)(&Kbuf[u][krow * 64 + kseg * 8]);
    vlo[u] = (unsigned)(uintptr_t)(&Vbuf[u][vrow * 32 + vseg * 8]);
  }

  v16bf aq0 = loadA_bf16(Qb + (size_t)(q0 + ln) * DV + 0, lane);
  v16bf aq1 = loadA_bf16(Qb + (size_t)(q0 + ln) * DV + 32, lane);
  v8f o0 = {}, o1 = {}, o2 = {}, o3 = {};
  float mrow[8], srow[8];
#pragma unroll
  for (int j = 0; j < 8; ++j) { mrow[j] = -1e30f; srow[j] = 0.f; }
  const float scale = 0.044194173824159216f;  // 1/sqrt(DIM_V=512)

  // prologue: stage chunk 0 into buffer 0
  async_g2l_b128(klo[0], kg);
  async_g2l_b128(vlo[0], vg);
  wait_async0();
  __syncthreads();

  for (int kc = 0; kc < SEQ; kc += 32) {
    int buf = (kc >> 5) & 1;
    if (kc + 32 < SEQ) {  // prefetch next chunk into other buffer
      async_g2l_b128(klo[buf ^ 1], kg + (size_t)(kc + 32) * DV);
      async_g2l_b128(vlo[buf ^ 1], vg + (kc + 32));
    }
    const unsigned short* Kc = &Kbuf[buf][0];
    const unsigned short* Vc = &Vbuf[buf][0];

    v16bf bk0 = loadB_bf16(Kc + (size_t)ln * 64 + 0, lane);
    v16bf bk1 = loadB_bf16(Kc + (size_t)ln * 64 + 32, lane);
    v16bf bk2 = loadB_bf16(Kc + (size_t)(16 + ln) * 64 + 0, lane);
    v16bf bk3 = loadB_bf16(Kc + (size_t)(16 + ln) * 64 + 32, lane);
    v8f s0 = {}, s1 = {};
    s0 = wmma_bf16(aq0, bk0, s0);
    s0 = wmma_bf16(aq1, bk1, s0);
    s1 = wmma_bf16(aq0, bk2, s1);
    s1 = wmma_bf16(aq1, bk3, s1);
#pragma unroll
    for (int j = 0; j < 8; ++j) {
      float v0 = s0[j] * scale, v1 = s1[j] * scale;
      float mx = fmaxf(v0, v1);
      mx = fmaxf(mx, __shfl_xor(mx, 8, 32));
      mx = fmaxf(mx, __shfl_xor(mx, 4, 32));
      mx = fmaxf(mx, __shfl_xor(mx, 2, 32));
      mx = fmaxf(mx, __shfl_xor(mx, 1, 32));
      float mnew = fmaxf(mrow[j], mx);
      float corr = __expf(mrow[j] - mnew);
      float p0 = __expf(v0 - mnew), p1 = __expf(v1 - mnew);
      float rs = p0 + p1;
      rs += __shfl_xor(rs, 8, 32);
      rs += __shfl_xor(rs, 4, 32);
      rs += __shfl_xor(rs, 2, 32);
      rs += __shfl_xor(rs, 1, 32);
      srow[j] = srow[j] * corr + rs;
      mrow[j] = mnew;
      o0[j] *= corr; o1[j] *= corr; o2[j] *= corr; o3[j] *= corr;
      s0[j] = p0; s1[j] = p1;
    }
    // repack P (C layout f32) -> row-major bf16 in LDS, reload as A operand
#pragma unroll
    for (int j = 0; j < 8; ++j) {
      int r = j + lh * 8;
      myl[r * 32 + ln] = f2bf(s0[j]);
      myl[r * 32 + 16 + ln] = f2bf(s1[j]);
    }
    v16bf ap = loadA_bf16(myl + ln * 32, lane);
    v16bf bv0 = loadB_bf16(Vc + (size_t)(0 + ln) * 32, lane);
    v16bf bv1 = loadB_bf16(Vc + (size_t)(16 + ln) * 32, lane);
    v16bf bv2 = loadB_bf16(Vc + (size_t)(32 + ln) * 32, lane);
    v16bf bv3 = loadB_bf16(Vc + (size_t)(48 + ln) * 32, lane);
    o0 = wmma_bf16(ap, bv0, o0);
    o1 = wmma_bf16(ap, bv1, o1);
    o2 = wmma_bf16(ap, bv2, o2);
    o3 = wmma_bf16(ap, bv3, o3);

    wait_async0();     // next buffer landed (per-wave), readers of cur done
    __syncthreads();   // make staged data visible block-wide
  }
  v8f oc[4] = {o0, o1, o2, o3};
  float* Ob = Opre + ((size_t)b * SEQ + q0) * DV + h * DH;
#pragma unroll
  for (int t = 0; t < 4; ++t) {
    int nb = t * 16 + ln;
#pragma unroll
    for (int j = 0; j < 8; ++j) {
      int r = j + lh * 8;
      float q = bf2f(Qb[(size_t)(q0 + r) * DV + nb]);
      Ob[(size_t)r * DV + nb] = oc[t][j] / srow[j] + q;
    }
  }
}

// ---- LayerNorm (wave per row, d=512) --------------------------------------
__global__ void __launch_bounds__(256)
mab_ln(const float* __restrict__ X, const float* __restrict__ g,
       const float* __restrict__ be, float* __restrict__ Yf,
       unsigned short* __restrict__ Yb, int M) {
  int lane = threadIdx.x & 31, wave = threadIdx.x >> 5;
  int row = blockIdx.x * 8 + wave;
  if (row >= M) return;
  const float* x = X + (size_t)row * DV;
  float vals[16];
  float s = 0.f;
#pragma unroll
  for (int i = 0; i < 16; ++i) { vals[i] = x[i * 32 + lane]; s += vals[i]; }
  s += __shfl_xor(s, 16, 32); s += __shfl_xor(s, 8, 32);
  s += __shfl_xor(s, 4, 32);  s += __shfl_xor(s, 2, 32); s += __shfl_xor(s, 1, 32);
  float mu = s * (1.0f / DV);
  float v = 0.f;
#pragma unroll
  for (int i = 0; i < 16; ++i) { float d = vals[i] - mu; v += d * d; }
  v += __shfl_xor(v, 16, 32); v += __shfl_xor(v, 8, 32);
  v += __shfl_xor(v, 4, 32);  v += __shfl_xor(v, 2, 32); v += __shfl_xor(v, 1, 32);
  float rstd = rsqrtf(v * (1.0f / DV) + 1e-5f);
#pragma unroll
  for (int i = 0; i < 16; ++i) {
    int col = i * 32 + lane;
    float y = (vals[i] - mu) * rstd * g[col] + be[col];
    if (Yf) Yf[(size_t)row * DV + col] = y;
    if (Yb) Yb[(size_t)row * DV + col] = f2bf(y);
  }
}

// ---- FFN GEMM: out = R + relu(A(bf16) @ W(bf16)^T + bias), f32 out --------
__global__ void __launch_bounds__(256)
mab_gemm_ffn(const unsigned short* __restrict__ A, const unsigned short* __restrict__ W,
             const float* __restrict__ bias, const float* __restrict__ R,
             float* __restrict__ out, int M) {
  int lane = threadIdx.x & 31, wave = threadIdx.x >> 5;
  int wid = blockIdx.x * 8 + wave;
  const int ntiles = DV / 64;  // 8
  int m0 = (wid / ntiles) * 32, n0 = (wid % ntiles) * 64;
  if (m0 >= M) return;
  int ln = lane & 15, lh = lane >> 4;
  v8f c[2][4];
#pragma unroll
  for (int i = 0; i < 2; ++i)
#pragma unroll
    for (int t = 0; t < 4; ++t) c[i][t] = {};
  for (int k0 = 0; k0 < DV; k0 += 32) {
    v16bf a0 = loadA_bf16(A + (size_t)(m0 + ln) * DV + k0, lane);
    v16bf a1 = loadA_bf16(A + (size_t)(m0 + 16 + ln) * DV + k0, lane);
    v16bf b0 = loadB_bf16(W + (size_t)(n0 + ln) * DV + k0, lane);
    v16bf b1 = loadB_bf16(W + (size_t)(n0 + 16 + ln) * DV + k0, lane);
    v16bf b2 = loadB_bf16(W + (size_t)(n0 + 32 + ln) * DV + k0, lane);
    v16bf b3 = loadB_bf16(W + (size_t)(n0 + 48 + ln) * DV + k0, lane);
    c[0][0] = wmma_bf16(a0, b0, c[0][0]);
    c[0][1] = wmma_bf16(a0, b1, c[0][1]);
    c[0][2] = wmma_bf16(a0, b2, c[0][2]);
    c[0][3] = wmma_bf16(a0, b3, c[0][3]);
    c[1][0] = wmma_bf16(a1, b0, c[1][0]);
    c[1][1] = wmma_bf16(a1, b1, c[1][1]);
    c[1][2] = wmma_bf16(a1, b2, c[1][2]);
    c[1][3] = wmma_bf16(a1, b3, c[1][3]);
  }
#pragma unroll
  for (int i = 0; i < 2; ++i) {
#pragma unroll
    for (int t = 0; t < 4; ++t) {
      int mb = m0 + i * 16, nb = n0 + t * 16 + ln;
      float bi = bias[nb];
#pragma unroll
      for (int j = 0; j < 8; ++j) {
        int m = mb + j + lh * 8;
        float v = fmaxf(c[i][t][j] + bi, 0.0f);
        size_t idx = (size_t)m * DV + nb;
        out[idx] = R[idx] + v;
      }
    }
  }
}

extern "C" void kernel_launch(void* const* d_in, const int* in_sizes, int n_in,
                              void* d_out, int out_size, void* d_ws, size_t ws_size,
                              hipStream_t stream) {
  (void)in_sizes; (void)n_in; (void)out_size; (void)ws_size;
  const float* Q   = (const float*)d_in[0];
  const float* K   = (const float*)d_in[1];
  const float* Wq  = (const float*)d_in[2];
  const float* bq  = (const float*)d_in[3];
  const float* Wk  = (const float*)d_in[4];
  const float* bk  = (const float*)d_in[5];
  const float* Wv  = (const float*)d_in[6];
  const float* bv  = (const float*)d_in[7];
  const float* Wo  = (const float*)d_in[8];
  const float* bo  = (const float*)d_in[9];
  const float* g0  = (const float*)d_in[10];
  const float* be0 = (const float*)d_in[11];
  const float* g1  = (const float*)d_in[12];
  const float* be1 = (const float*)d_in[13];
  float* out = (float*)d_out;

  const int Mrows = 8 * SEQ;                 // 8192
  const size_t elems = (size_t)Mrows * DV;   // 4,194,304
  const size_t wel = (size_t)DV * DV;        // 262,144
  char* ws = (char*)d_ws;
  unsigned short* Qbf = (unsigned short*)ws;      // dead after projections
  unsigned short* Kbf = Qbf + elems;              // dead after projections
  unsigned short* Qp  = Kbf + elems;              // dead after attention
  unsigned short* Kp  = Qp + elems;               // dead after attention
  unsigned short* VpT = Kp + elems;
  unsigned short* Wqb = VpT + elems;
  unsigned short* Wkb = Wqb + wel;
  unsigned short* Wvb = Wkb + wel;
  unsigned short* Wob = Wvb + wel;
  float* O1f = (float*)(Wob + wel);
  unsigned short* O1b = (unsigned short*)(O1f + elems);
  float* Opre = (float*)ws;        // alias Qbf+Kbf (exactly elems floats)
  float* O2   = (float*)Qp;        // alias Qp+Kp  (exactly elems floats)

  dim3 blk(256);
  // fp32 -> bf16 pre-conversion (once, out of all GEMM hot loops)
  mab_cvt_bf16<<<1024, blk, 0, stream>>>(Q, Qbf, (int)(elems / 8));
  mab_cvt_bf16<<<1024, blk, 0, stream>>>(K, Kbf, (int)(elems / 8));
  mab_cvt_bf16<<<128, blk, 0, stream>>>(Wq, Wqb, (int)(wel / 8));
  mab_cvt_bf16<<<128, blk, 0, stream>>>(Wk, Wkb, (int)(wel / 8));
  mab_cvt_bf16<<<128, blk, 0, stream>>>(Wv, Wvb, (int)(wel / 8));
  mab_cvt_bf16<<<128, blk, 0, stream>>>(Wo, Wob, (int)(wel / 8));

  int gemm_blocks = (Mrows / 32) * (DV / 64) / 8;  // 256
  mab_gemm_proj<<<gemm_blocks, blk, 0, stream>>>(Qbf, Wqb, bq, Qp, Mrows, 0);
  mab_gemm_proj<<<gemm_blocks, blk, 0, stream>>>(Kbf, Wkb, bk, Kp, Mrows, 0);
  mab_gemm_proj<<<gemm_blocks, blk, 0, stream>>>(Kbf, Wvb, bv, VpT, Mrows, 1);
  mab_attn<<<512, blk, 0, stream>>>(Qp, Kp, VpT, Opre);
  mab_ln<<<Mrows / 8, blk, 0, stream>>>(Opre, g0, be0, O1f, O1b, Mrows);
  mab_gemm_ffn<<<gemm_blocks, blk, 0, stream>>>(O1b, Wob, bo, O1f, O2, Mrows);
  mab_ln<<<Mrows / 8, blk, 0, stream>>>(O2, g1, be1, out, (unsigned short*)nullptr, Mrows);
}
